// TrilinearDistance_10385230922579
// MI455X (gfx1250) — matrix operand order repaired and checked by reading, not compiled, not run on hardware
//
#include <hip/hip_runtime.h>
#include <hip/hip_bf16.h>

typedef _Float16 half_t;
typedef __attribute__((ext_vector_type(16))) _Float16 v16h;
typedef __attribute__((ext_vector_type(8)))  _Float16 v8h;
typedef __attribute__((ext_vector_type(8)))  float    v8f;
typedef uint32_t u32x4 __attribute__((ext_vector_type(4)));
typedef uint32_t u32x8 __attribute__((ext_vector_type(8)));

#define BB 4
#define SS 256
#define TT 256
#define LDm 256
#define RDm 256
#define AD 128

// ---------- WMMA fragment loaders (wave32, 16-bit, per ISA 7.12.2) ----------
// A (16x32, M x K), source row-major [M][K] with row stride ld (halves):
//   lane L: M = L&15 ; lanes 0-15 hold K=0..7 (V0-3) and K=16..23 (V4-7),
//           lanes 16-31 hold K=8..15 and K=24..31.
__device__ __forceinline__ v16h frag_a(const half_t* base, int ld, int lane) {
  const int m = lane & 15, sel = lane >> 4;
  const half_t* p = base + (size_t)m * ld + sel * 8;
  union { v16h v; v8h h[2]; } u;
  u.h[0] = *(const v8h*)(p);        // K = sel*8 + 0..7
  u.h[1] = *(const v8h*)(p + 16);   // K = sel*8 + 16..23
  return u.v;
}
// B (32x16, K x N), source stored N-major [N][K] with row stride ld (halves):
//   lane L: N = L&15 ; lanes 0-15 hold K=0..7 and K=16..23,
//           lanes 16-31 hold K=8..15 and K=24..31.
__device__ __forceinline__ v16h frag_b(const half_t* base, int ld, int lane) {
  const int n = lane & 15, sel = lane >> 4;
  const half_t* p = base + (size_t)n * ld + sel * 8;
  union { v16h v; v8h h[2]; } u;
  u.h[0] = *(const v8h*)(p);
  u.h[1] = *(const v8h*)(p + 16);
  return u.v;
}

// ---------- TDM: DMA one 2D tile of 16-bit data from global into LDS ----------
// elems0 = contiguous (innermost) length, elems1 = #rows, stride0 = row stride
// (all in 2-byte elements). Issued once per call; tracked by TENSORcnt.
__device__ __forceinline__ void tdm_load_2d_f16(uint32_t lds_addr,
                                                const void* gaddr,
                                                uint32_t elems0, uint32_t elems1,
                                                uint32_t stride0) {
  const uint64_t ga = (uint64_t)(uintptr_t)gaddr;
  u32x4 g0;
  g0[0] = 1u;                                               // count=1 (valid user D#)
  g0[1] = lds_addr;                                         // LDS byte address
  g0[2] = (uint32_t)ga;                                     // global_addr[31:0]
  g0[3] = (uint32_t)((ga >> 32) & 0x1FFFFFFu) | (2u << 30); // addr[56:32] | type=2
  u32x8 g1;
  g1[0] = (1u << 16);                                       // wg_mask=0, data_size=1 (2B)
  g1[1] = (elems0 & 0xFFFFu) << 16;                         // tensor_dim0[15:0]
  g1[2] = ((elems0 >> 16) & 0xFFFFu) | ((elems1 & 0xFFFFu) << 16); // dim0 hi | dim1 lo
  g1[3] = ((elems1 >> 16) & 0xFFFFu) | ((elems0 & 0xFFFFu) << 16); // dim1 hi | tile_dim0
  g1[4] = (elems1 & 0xFFFFu);                               // tile_dim1 ; tile_dim2=0
  g1[5] = stride0;                                          // tensor_dim0_stride[31:0]
  g1[6] = 0u;                                               // stride0 hi | dim1_stride lo
  g1[7] = 0u;
  asm volatile("tensor_load_to_lds %0, %1" :: "s"(g0), "s"(g1) : "memory");
}

// ---------- conversion kernels ----------
__global__ void cvt_f16_kernel(const float* __restrict__ src,
                               half_t* __restrict__ dst, int n) {
  int i = blockIdx.x * blockDim.x + threadIdx.x;
  if (i < n) dst[i] = (half_t)src[i];
}

// WT[a][r][l] = (f16) W[a][l][r]   (block = (a,r), thread = l; coalesced writes)
__global__ void cvtT_kernel(const float* __restrict__ W,
                            half_t* __restrict__ WT) {
  const int a = blockIdx.x >> 8;
  const int r = blockIdx.x & 255;
  const int l = threadIdx.x;
  WT[((size_t)a * RDm + r) * LDm + l] =
      (half_t)W[((size_t)a * LDm + l) * RDm + r];
}

// ---------- Stage A: G[b,a] = data[b] (SxL) @ W[a] (LxR), f16 out ----------
__global__ void stageA_kernel(const half_t* __restrict__ dataH,
                              const half_t* __restrict__ WTH,
                              half_t* __restrict__ G) {
  extern __shared__ half_t smem[];        // 256 KB: sA[SS][LDm], sB[RDm][LDm]
  half_t* sA = smem;
  half_t* sB = smem + SS * LDm;

  const int b = blockIdx.x / AD;
  const int a = blockIdx.x % AD;

  // Wave 0 DMAs both matrices into LDS via the Tensor Data Mover, waits for
  // TENSORcnt==0, then everyone syncs on the workgroup barrier.
  if (threadIdx.x < 32) {
    tdm_load_2d_f16((uint32_t)(uintptr_t)sA, dataH + (size_t)b * SS * LDm,
                    LDm, SS, LDm);
    tdm_load_2d_f16((uint32_t)(uintptr_t)sB, WTH + (size_t)a * RDm * LDm,
                    LDm, RDm, LDm);
    __builtin_amdgcn_s_wait_tensorcnt(0);
  }
  __syncthreads();

  const int wave = threadIdx.x >> 5;
  const int lane = threadIdx.x & 31;
  const int m = lane & 15, sel = lane >> 4;
  half_t* __restrict__ Gb = G + (size_t)blockIdx.x * SS * RDm;

  for (int rr = 0; rr < 2; ++rr) {
    const int col0 = (wave * 2 + rr) * 16;      // r-tile owned by this wave
    v16h bfr[8];
#pragma unroll
    for (int kk = 0; kk < 8; ++kk)
      bfr[kk] = frag_b(sB + (size_t)col0 * LDm + kk * 32, LDm, lane);

    for (int ts = 0; ts < 16; ++ts) {
      const int row0 = ts * 16;
      v8f acc = {};
#pragma unroll
      for (int kk = 0; kk < 8; ++kk) {
        v16h af = frag_a(sA + (size_t)row0 * LDm + kk * 32, LDm, lane);
        acc = __builtin_amdgcn_wmma_f32_16x16x32_f16(
            false, af, false, bfr[kk], (short)0, acc, false, false);
      }
#pragma unroll
      for (int v = 0; v < 8; ++v)
        Gb[(size_t)(row0 + v + 8 * sel) * RDm + col0 + m] = (half_t)acc[v];
    }
  }
}

// ---------- Stage B: out[b,s,t] = sum_a agg[a]*tanh( G[b,a,s,:] . crit[b,t,:] ) ----------
__global__ void stageB_kernel(const half_t* __restrict__ G,
                              const half_t* __restrict__ critH,
                              const float* __restrict__ agg,
                              float* __restrict__ out) {
  const int wave = threadIdx.x >> 5;
  const int lane = threadIdx.x & 31;
  const int pair = blockIdx.x * 8 + wave;       // 0..1023
  const int b    = pair >> 8;
  const int rem  = pair & 255;
  const int row0 = (rem >> 4) << 4;             // s-tile origin
  const int col0 = (rem & 15) << 4;             // t-tile origin
  const int m = lane & 15, sel = lane >> 4;

  // crit fragments are invariant over the a-loop: preload 8 chunks (64 VGPRs)
  const half_t* cb = critH + (size_t)b * TT * RDm + (size_t)col0 * RDm;
  v16h bfr[8];
#pragma unroll
  for (int kk = 0; kk < 8; ++kk)
    bfr[kk] = frag_b(cb + kk * 32, RDm, lane);

  v8f acc = {};
  const half_t* Ga = G + (size_t)b * AD * SS * RDm + (size_t)row0 * RDm;
  for (int a = 0; a < AD; ++a) {
    const half_t* Gp = Ga + (size_t)a * SS * RDm;
    __builtin_prefetch(Gp + (size_t)SS * RDm, 0, 0);  // next a's tile rows
    v8f bil = {};
#pragma unroll
    for (int kk = 0; kk < 8; ++kk) {
      v16h af = frag_a(Gp + kk * 32, RDm, lane);
      bil = __builtin_amdgcn_wmma_f32_16x16x32_f16(
          false, af, false, bfr[kk], (short)0, bil, false, false);
    }
    const float ga = agg[a];
#pragma unroll
    for (int v = 0; v < 8; ++v) acc[v] += ga * tanhf(bil[v]);
  }

#pragma unroll
  for (int v = 0; v < 8; ++v)
    out[(size_t)b * SS * TT + (size_t)(row0 + v + 8 * sel) * TT + col0 + m] =
        acc[v];
}

extern "C" void kernel_launch(void* const* d_in, const int* in_sizes, int n_in,
                              void* d_out, int out_size, void* d_ws, size_t ws_size,
                              hipStream_t stream) {
  (void)in_sizes; (void)n_in; (void)out_size; (void)ws_size;
  const float* data = (const float*)d_in[0];   // [B,S,L]
  const float* crit = (const float*)d_in[1];   // [B,T,R]
  const float* W    = (const float*)d_in[2];   // [A,L,R]
  const float* agg  = (const float*)d_in[3];   // [A]
  float* out = (float*)d_out;                  // [B,S,T]

  // workspace layout (halves): dataH | critH | WT | G
  half_t* dataH = (half_t*)d_ws;                       // B*S*L
  half_t* critH = dataH + (size_t)BB * SS * LDm;       // B*T*R
  half_t* WTH   = critH + (size_t)BB * TT * RDm;       // A*R*L (W transposed)
  half_t* G     = WTH + (size_t)AD * RDm * LDm;        // B*A*S*R

  const int nData = BB * SS * LDm;                     // 262144
  const int nCrit = BB * TT * RDm;                     // 262144

  cvt_f16_kernel<<<nData / 256, 256, 0, stream>>>(data, dataH, nData);
  cvt_f16_kernel<<<nCrit / 256, 256, 0, stream>>>(crit, critH, nCrit);
  cvtT_kernel<<<AD * RDm, LDm, 0, stream>>>(W, WTH);

  // Stage A: one block per (b,a); 256 KB dynamic LDS (data[b] + WT[a]) via TDM
  stageA_kernel<<<BB * AD, 256, 2 * SS * LDm * (int)sizeof(half_t), stream>>>(
      dataH, WTH, G);

  // Stage B: one wave per 16x16 (s,t) tile; 1024 tiles / 8 waves per block
  stageB_kernel<<<BB * (SS / 16) * (TT / 16) / 8, 256, 0, stream>>>(
      G, critH, agg, out);
}
